// MyJoiner_52089363366262
// MI455X (gfx1250) — matrix operand order repaired and checked by reading, not compiled, not run on hardware
//
#include <hip/hip_runtime.h>
#include <math.h>

// Problem constants (match reference)
#define B_ 4
#define T_ 512
#define U_ 128
#define D_ 512
#define V_ 1024

// WMMA vector types (gfx1250 CDNA5: v_wmma_f32_16x16x32_bf16)
typedef __attribute__((ext_vector_type(16))) __bf16 v16bf;
typedef __attribute__((ext_vector_type(8)))  float  v8f;

// ---- LDS geometry -----------------------------------------------------------
// A tile: full joint rows for one (b,t): 128 (u) x 512 (k) bf16, +8 pad elems
// so 16 lanes doing ds_load_b128 with row stride 1040B land on distinct banks.
#define A_PAD    8
#define A_STRIDE (D_ + A_PAD)              // 520 ushorts/row, 1040 B
#define A_BYTES  (U_ * A_STRIDE * 2)       // 133120 B
// B tile: 128 (n) x 32 (k) bf16, +8 pad, double buffered.
#define BK       32
#define B_PAD    8
#define B_STRIDE (BK + B_PAD)              // 40 ushorts/row, 80 B
#define B_TILE_N 128
#define B_BYTES  (B_TILE_N * B_STRIDE * 2) // 10240 B = 640 uint4 = 80 lines
#define LDS_BYTES (A_BYTES + 2 * B_BYTES)  // 153600 B (< 320 KB WGP spec)

#define NCHUNKS  (V_ / B_TILE_N)           // 8
#define KTILES   (D_ / BK)                 // 16
#define TILE_USH (B_TILE_N * B_STRIDE)     // 5120 ushorts per [nc][kt] tile
// Pre-converted, pre-swizzled bf16 W in workspace: [nc][kt][n][B_STRIDE]
#define W16_USHORTS ((size_t)NCHUNKS * KTILES * TILE_USH)
#define W16_BYTES   (W16_USHORTS * 2)      // 1,310,720 B

__device__ __forceinline__ unsigned short f2bf(float f) {
  // round-to-nearest-even f32 -> bf16
  unsigned int u = __builtin_bit_cast(unsigned int, f);
  u += 0x7FFFu + ((u >> 16) & 1u);
  return (unsigned short)(u >> 16);
}

__device__ __forceinline__ uint2 pack4(float a, float b, float c, float d) {
  uint2 r;
  r.x = (unsigned)f2bf(a) | ((unsigned)f2bf(b) << 16);
  r.y = (unsigned)f2bf(c) | ((unsigned)f2bf(d) << 16);
  return r;
}

__device__ __forceinline__ float gelu_exact(float x) {
  // approximate=False: 0.5*x*(1+erf(x/sqrt(2)))
  return 0.5f * x * (1.0f + erff(x * 0.70710678118654752f));
}

// WGP-scope prefetch (ISA 10.5: SCOPE 0 pulls into ALL cache levels on miss).
// __builtin_prefetch lowers to scope:SCOPE_SE, which skips the WGP cache, so
// emit the instruction directly; default scope field (unprinted) is 0 = WGP.
__device__ __forceinline__ void prefetch_wgp(const void* p) {
  asm volatile("global_prefetch_b8 %0, off" :: "v"(p));
}

union Frag { uint4 u4[2]; v16bf v; };

// One-time prep: W (V,D) f32 -> bf16, swizzled into LDS tile order so the main
// kernel's B-staging is a pure contiguous b128 copy (no per-use conversion).
__global__ void __launch_bounds__(256)
w_to_bf16_swizzled(const float* __restrict__ W, unsigned short* __restrict__ W16) {
  const int g = (blockIdx.x * 256 + threadIdx.x) * 4;  // element over V_*D_
  const int v = g >> 9;                                // / D_
  const int k = g & (D_ - 1);
  float4 wv = *(const float4*)(W + (size_t)v * D_ + k);
  const int nc = v >> 7, n = v & (B_TILE_N - 1);
  const int kt = k >> 5, kk = k & (BK - 1);
  const size_t dst = ((size_t)(nc * KTILES + kt) * B_TILE_N + n) * B_STRIDE + kk;
  *(uint2*)(W16 + dst) = pack4(wv.x, wv.y, wv.z, wv.w);
}

// Direct tile copy ws -> LDS: 640 uint4 slots; threads own {tid, tid+256},
// threads < 128 also own tid+512.  Loads should be WGP-cache hits thanks to
// the global_prefetch issued one K-step earlier, so the loadcnt wait is short.
__device__ __forceinline__ void copyB(unsigned short* sb,
                                      const unsigned short* wsTile, int tid) {
  const uint4* s = (const uint4*)wsTile;
  uint4* d = (uint4*)sb;
  d[tid]       = s[tid];
  d[tid + 256] = s[tid + 256];
  if (tid < 128) d[tid + 512] = s[tid + 512];
}

// Fallback staging (only if ws too small): convert f32 W on the fly.
__device__ __forceinline__ void stageB_convert(unsigned short* sb,
                                               const float* Wt, int tid) {
  for (int e = tid; e < B_TILE_N * (BK / 4); e += 256) {
    const int n = e >> 3, kk = (e & 7) << 2;
    float4 wv = *(const float4*)(Wt + (size_t)n * D_ + kk);
    *(uint2*)(sb + n * B_STRIDE + kk) = pack4(wv.x, wv.y, wv.z, wv.w);
  }
}

// One workgroup per (b,t): computes out[b,t,0:128,0:1024].
// GEMM: M=U (u), N=V (v), K=D (d); A built once with exact gelu, bf16 WMMA,
// f32 accumulate, nontemporal output stores (1 GiB stream must not evict W/L2).
template <bool PRECONV>
__global__ void __launch_bounds__(256)
joiner_wmma(const float* __restrict__ src,
            const float* __restrict__ tgt,
            const float* __restrict__ W,
            const unsigned short* __restrict__ W16,
            const float* __restrict__ bias,
            float* __restrict__ out) {
  extern __shared__ char smem[];
  unsigned short* sA  = (unsigned short*)smem;
  unsigned short* sB0 = (unsigned short*)(smem + A_BYTES);

  const int tid  = threadIdx.x;
  const int lane = tid & 31;       // wave32
  const int l16  = lane & 15;
  const int half = lane >> 4;
  const int wave = tid >> 5;       // 0..7
  const int wm   = wave & 3;       // 4 waves along M
  const int wn   = wave >> 2;      // 2 waves along N
  const int bIdx = blockIdx.x / T_;
  const int tIdx = blockIdx.x % T_;

  const float* srcRow = src + ((size_t)bIdx * T_ + tIdx) * D_;
  const float* tgtB   = tgt + (size_t)bIdx * U_ * D_;

  // ---- Stage A = bf16(gelu(src_row + tgt_row)) : 128 x 512, once per WG ----
  for (int e = tid; e < U_ * (D_ / 4); e += 256) {
    int u = e >> 7;              // D_/4 == 128 vec4 slots per row
    int k = (e & 127) << 2;
    float4 tv = *(const float4*)(tgtB + (size_t)u * D_ + k);
    float4 sv = *(const float4*)(srcRow + k);
    *(uint2*)(sA + u * A_STRIDE + k) =
        pack4(gelu_exact(sv.x + tv.x), gelu_exact(sv.y + tv.y),
              gelu_exact(sv.z + tv.z), gelu_exact(sv.w + tv.w));
  }

  const int wmBase = wm * 32;     // wave M tile: 32 rows (2 frags)
  const int wnBase = wn * 64;     // wave N tile: 64 cols (4 frags)
  float* outBase = out + (((size_t)bIdx * T_ + tIdx) * U_) * V_;

  for (int nc = 0; nc < NCHUNKS; ++nc) {
    const int vBase = nc * B_TILE_N;
    const unsigned short* wsNc =
        PRECONV ? (W16 + (size_t)nc * KTILES * TILE_USH) : nullptr;

    // Prologue: stage W k-tile 0 into buffer 0.
    if (PRECONV)
      copyB(sB0, wsNc, tid);
    else
      stageB_convert(sB0, W + (size_t)vBase * D_, tid);

    v8f acc[2][4];
#pragma unroll
    for (int fm = 0; fm < 2; ++fm)
#pragma unroll
      for (int fn = 0; fn < 4; ++fn)
        acc[fm][fn] = (v8f){0.f, 0.f, 0.f, 0.f, 0.f, 0.f, 0.f, 0.f};

#pragma unroll 2
    for (int kt = 0; kt < KTILES; ++kt) {
      __syncthreads();  // buffer (kt&1) ready; prior reads of (kt+1)&1 done

      // Prefetch (WGP scope) the tile we will copy AFTER this step's WMMAs:
      // next k-tile, or the next N-chunk's tile 0 (contiguous in the swizzled
      // layout).
      if (PRECONV) {
        const unsigned short* pf = nullptr;
        if (kt + 1 < KTILES)        pf = wsNc + (size_t)(kt + 1) * TILE_USH;
        else if (nc + 1 < NCHUNKS)  pf = wsNc + (size_t)KTILES * TILE_USH;
        if (pf && tid < B_BYTES / 128)  // 80 cache lines of 128 B
          prefetch_wgp((const char*)pf + tid * 128);
      }

      // ---- compute on current buffer ----
      const unsigned short* sb = sB0 + (kt & 1) * (B_BYTES / 2);
      const int kBase = kt * BK;

      Frag a[2], bf[4];
#pragma unroll
      for (int fm = 0; fm < 2; ++fm) {
        // A 16-bit 16x32 layout: lanes0-15 K=[0..7],[16..23]; lanes16-31 +8
        const unsigned short* ap =
            sA + (wmBase + fm * 16 + l16) * A_STRIDE + kBase + half * 8;
        a[fm].u4[0] = *(const uint4*)ap;
        a[fm].u4[1] = *(const uint4*)(ap + 16);
      }
#pragma unroll
      for (int fn = 0; fn < 4; ++fn) {
        // B 16-bit 32x16 layout: lane = column n, lanes0-15 K=[0..7],[16..23];
        // lanes16-31 +8 (same K order as A)
        const unsigned short* bp =
            sb + (wnBase + fn * 16 + l16) * B_STRIDE + half * 8;
        bf[fn].u4[0] = *(const uint4*)bp;
        bf[fn].u4[1] = *(const uint4*)(bp + 16);
      }
#pragma unroll
      for (int fm = 0; fm < 2; ++fm)
#pragma unroll
        for (int fn = 0; fn < 4; ++fn)
          acc[fm][fn] = __builtin_amdgcn_wmma_f32_16x16x32_bf16(
              false, a[fm].v, false, bf[fn].v, (short)0, acc[fm][fn],
              false, false);

      // Stage next k-tile into the other buffer (loads hit the WGP cache via
      // prefetch; no long-lived registers across the WMMA block -> no spills).
      if (kt + 1 < KTILES) {
        unsigned short* sbn = sB0 + ((kt + 1) & 1) * (B_BYTES / 2);
        if (PRECONV)
          copyB(sbn, wsNc + (size_t)(kt + 1) * TILE_USH, tid);
        else
          stageB_convert(sbn, W + (size_t)vBase * D_ + (kt + 1) * BK, tid);
      }
    }

    // ---- epilogue: bias + nontemporal streaming stores ----
    float bv[4];
#pragma unroll
    for (int fn = 0; fn < 4; ++fn)
      bv[fn] = bias[vBase + wnBase + fn * 16 + l16];
#pragma unroll
    for (int fm = 0; fm < 2; ++fm) {
#pragma unroll
      for (int fn = 0; fn < 4; ++fn) {
        const int v = vBase + wnBase + fn * 16 + l16;
#pragma unroll
        for (int r = 0; r < 8; ++r) {
          // C/D layout: VGPR r -> M = r (lanes 0-15) or r+8 (lanes 16-31)
          const int m = wmBase + fm * 16 + half * 8 + r;
          __builtin_nontemporal_store(acc[fm][fn][r] + bv[fn],
                                      outBase + (size_t)m * V_ + v);
        }
      }
    }
    // Next nc's prologue writes buf0, which all waves stopped reading before
    // this chunk's last barrier -> no trailing barrier needed.
  }
}

// Reference returns (output, source_lengths, target_lengths): append the two
// length vectors (int32 bits) after the B*T*U*V floats.
__global__ void tail_copy(const int* __restrict__ sl, const int* __restrict__ tl,
                          int* __restrict__ dst) {
  int i = threadIdx.x;
  if (i < B_)           dst[i] = sl[i];
  else if (i < 2 * B_)  dst[i] = tl[i - B_];
}

extern "C" void kernel_launch(void* const* d_in, const int* in_sizes, int n_in,
                              void* d_out, int out_size, void* d_ws, size_t ws_size,
                              hipStream_t stream) {
  (void)in_sizes; (void)n_in; (void)out_size;
  const float* src  = (const float*)d_in[0];
  const int*   slen = (const int*)d_in[1];
  const float* tgt  = (const float*)d_in[2];
  const int*   tlen = (const int*)d_in[3];
  const float* W    = (const float*)d_in[4];
  const float* bias = (const float*)d_in[5];
  float* out = (float*)d_out;

  if (ws_size >= W16_BYTES) {
    unsigned short* W16 = (unsigned short*)d_ws;
    // 524288 elements / 4 per thread / 256 threads = 512 blocks
    w_to_bf16_swizzled<<<dim3((V_ * D_) / 4 / 256), dim3(256), 0, stream>>>(W, W16);
    joiner_wmma<true><<<dim3(B_ * T_), dim3(256), LDS_BYTES, stream>>>(
        src, tgt, W, W16, bias, out);
  } else {
    joiner_wmma<false><<<dim3(B_ * T_), dim3(256), LDS_BYTES, stream>>>(
        src, tgt, W, nullptr, bias, out);
  }

  tail_copy<<<dim3(1), dim3(2 * B_), 0, stream>>>(
      slen, tlen, (int*)d_out + (size_t)B_ * T_ * U_ * V_);
}
